// MultiHeadAttention_32358283608293
// MI455X (gfx1250) — matrix operand order repaired and checked by reading, not compiled, not run on hardware
//
#include <hip/hip_runtime.h>

#define DEV __device__ __forceinline__

typedef __attribute__((ext_vector_type(16))) __bf16 bf16x16;
typedef __attribute__((ext_vector_type(8)))  __bf16 bf16x8;
typedef __attribute__((ext_vector_type(8)))  float  f32x8;
typedef __attribute__((ext_vector_type(8)))  unsigned short u16x8;

static constexpr int Bb = 8, Ss = 1024, Ee = 768, Hh = 12, Dd = 64;

// ---------- helpers ----------
DEV unsigned short f2bf(float f) {
  union { float f; unsigned u; } x; x.f = f;
  unsigned r = x.u + 0x7FFFu + ((x.u >> 16) & 1u);   // round-to-nearest-even
  return (unsigned short)(r >> 16);
}
DEV __bf16 f2bfh(float f) { return (__bf16)f; }      // native cvt on gfx1250

// lane ^ 16 exchange without an index VGPR: ds_swizzle SWAPX16 (xor=0x10,and=0x1f)
DEV float swap16(float x) {
  int i = __builtin_bit_cast(int, x);
  i = __builtin_amdgcn_ds_swizzle(i, 0x401F);
  return __builtin_bit_cast(float, i);
}

DEV f32x8 wmma_bf16(bf16x16 a, bf16x16 b, f32x8 c) {
  // (neg_a, A, neg_b, B, c_mod, C, reuse_a, reuse_b)
  return __builtin_amdgcn_wmma_f32_16x16x32_bf16(false, a, false, b, (short)0, c, false, false);
}

// A-fragment (16x32 bf16, MxK), A row-major, leading dim lda (elements).
// lanes 0-15: M=lane, K = k0..k0+15; lanes 16-31: K = k0+16..k0+31.
DEV bf16x16 load_frag_a(const unsigned short* A, int lda, int m0, int k0, int lane) {
  int m  = m0 + (lane & 15);
  int kb = k0 + ((lane >> 4) << 4);
  const bf16x8* p0 = reinterpret_cast<const bf16x8*>(A + (long)m * lda + kb);
  const bf16x8* p1 = reinterpret_cast<const bf16x8*>(A + (long)m * lda + kb + 8);
  bf16x8 lo = *p0, hi = *p1;
  bf16x16 r;
#pragma unroll
  for (int i = 0; i < 8; i++) { r[i] = lo[i]; r[i + 8] = hi[i]; }
  return r;
}

// B-fragment (32x16 bf16, KxN) from an (N,K) row-major "Bt" matrix (Bt[n][k] = B[k][n]).
// lanes 0-15: N=n0+lane, K = k0..k0+15; lanes 16-31: K = k0+16..k0+31.
DEV bf16x16 load_frag_b(const unsigned short* Bt, int ldb, int n0, int k0, int lane) {
  int n  = n0 + (lane & 15);
  int kb = k0 + ((lane >> 4) << 4);
  const bf16x8* p = reinterpret_cast<const bf16x8*>(Bt + (long)n * ldb + kb);
  bf16x8 lo = p[0], hi = p[1];
  bf16x16 r;
#pragma unroll
  for (int i = 0; i < 8; i++) { r[i] = lo[i]; r[i + 8] = hi[i]; }
  return r;
}

// ---------- conversion kernels ----------
__global__ void k_f32_to_bf16(const float* __restrict__ in, unsigned short* __restrict__ out, long n) {
  long i = (long)blockIdx.x * blockDim.x + threadIdx.x;
  long stride = (long)gridDim.x * blockDim.x;
  for (; i < n; i += stride) out[i] = f2bf(in[i]);
}

// in: (nMat, R, C) fp32 -> out: (nMat, C, R) bf16 ; nMat = gridDim.z
__global__ void k_transpose_bf16(const float* __restrict__ in, unsigned short* __restrict__ out,
                                 int R, int C) {
  long mat = blockIdx.z;
  const float* src = in + mat * (long)R * C;
  unsigned short* dst = out + mat * (long)R * C;
  int total = R * C;
  for (int idx = blockIdx.x * blockDim.x + threadIdx.x; idx < total;
       idx += gridDim.x * blockDim.x) {
    int r = idx / C, c = idx % C;
    dst[(long)c * R + r] = f2bf(src[idx]);
  }
}

// ---------- generic WMMA GEMM:  C(MxN) = scale*(A(MxK) @ BtT + bias) ----------
// wave tile = 32x64 (2 A-frags x 4 B-frags -> 8 WMMA/k-step, B reused 2x).
// workgroup = 8 waves -> 256x64 per block.
// MODE 0: bf16 row-major out | MODE 1: bf16 transposed out (out[n*ldc+m]) | MODE 2: fp32 row-major out
template <int MODE>
__global__ void __launch_bounds__(256)
k_gemm(const unsigned short* __restrict__ A, int lda, long aStride, int hDiv,
       const unsigned short* __restrict__ Bt, int ldb, long bStride,
       const float* __restrict__ bias, int biasStride,
       void* __restrict__ Cout, int ldc, long cStride, int K, float outScale) {
  int z = blockIdx.z;
  int bi = z / hDiv, hi = z % hDiv;
  A  += (long)bi * aStride;
  Bt += (long)hi * bStride;
  const float* bp = bias ? bias + (long)hi * biasStride : nullptr;

  int lane = threadIdx.x & 31;
  int wave = threadIdx.x >> 5;
  int m0 = blockIdx.x * 256 + wave * 32;
  int n0 = blockIdx.y * 64;

  f32x8 acc[2][4] = {};
  for (int k0 = 0; k0 < K; k0 += 32) {
    bf16x16 a0 = load_frag_a(A, lda, m0,      k0, lane);
    bf16x16 a1 = load_frag_a(A, lda, m0 + 16, k0, lane);
#pragma unroll
    for (int t = 0; t < 4; t++) {
      bf16x16 b = load_frag_b(Bt, ldb, n0 + t * 16, k0, lane);
      acc[0][t] = wmma_bf16(a0, b, acc[0][t]);
      acc[1][t] = wmma_bf16(a1, b, acc[1][t]);
    }
  }

  int half = lane >> 4, col = lane & 15;
#pragma unroll
  for (int mi = 0; mi < 2; mi++) {
    int mBase = m0 + mi * 16;
#pragma unroll
    for (int t = 0; t < 4; t++) {
      int n = n0 + t * 16 + col;
      float bv = bp ? bp[n] : 0.f;
      if (MODE == 0) {
        unsigned short* O = (unsigned short*)Cout + (long)z * cStride;
#pragma unroll
        for (int r = 0; r < 8; r++) {
          int m = mBase + half * 8 + r;
          O[(long)m * ldc + n] = f2bf((acc[mi][t][r] + bv) * outScale);
        }
      } else if (MODE == 1) {
        unsigned short* O = (unsigned short*)Cout + (long)z * cStride;
        u16x8 pk;
#pragma unroll
        for (int r = 0; r < 8; r++) pk[r] = f2bf((acc[mi][t][r] + bv) * outScale);
        *reinterpret_cast<u16x8*>(O + (long)n * ldc + mBase + half * 8) = pk;
      } else {
        float* O = (float*)Cout + (long)z * cStride;
#pragma unroll
        for (int r = 0; r < 8; r++) {
          int m = mBase + half * 8 + r;
          O[(long)m * ldc + n] = (acc[mi][t][r] + bv) * outScale;
        }
      }
    }
  }
}

// ---------- fused causal flash attention (transposed-score formulation) ----------
// S^T = K @ Q^T so the softmax key-axis lies along accumulator slots (in-lane),
// and O^T = V^T @ P^T so P^T feeds WMMA with only a half-wave swap (no LDS tile).
// grid (S/128, B*H), block 256. Each wave handles 16 query rows independently.
// Q,K: (B,H,S,D) bf16 row-major (Q pre-scaled by 1/sqrt(S)). Vt: (B,H,D,S) bf16.
__global__ void __launch_bounds__(256)
k_flash(const unsigned short* __restrict__ Q, const unsigned short* __restrict__ Kc,
        const unsigned short* __restrict__ Vt, unsigned short* __restrict__ Hid) {
  int z = blockIdx.y;
  int b = z / Hh, h = z % Hh;
  const unsigned short* q = Q  + (long)z * Ss * Dd;
  const unsigned short* k = Kc + (long)z * Ss * Dd;
  const unsigned short* v = Vt + (long)z * Dd * Ss;

  int lane = threadIdx.x & 31, wave = threadIdx.x >> 5;
  int half = lane >> 4, col = lane & 15;
  int mw = blockIdx.x * 128 + wave * 16;
  int m_col = mw + col;                 // this lane's query row (column of S^T)

  // Q as resident B-operand fragments (d = 0..31, 32..63)
  bf16x16 qb0 = load_frag_b(q, Dd, mw, 0, lane);
  bf16x16 qb1 = load_frag_b(q, Dd, mw, 32, lane);

  f32x8 o[4] = {};                      // O^T fragments: frag t = d rows t*16..t*16+15
  float mst = -1e30f, lst = 0.f;

  int nTB   = (mw + 15) / 32 + 1;       // key blocks touched under causality
  int nFull = (mw + 1) / 32;            // blocks entirely below the diagonal

  for (int tb = 0; tb < nTB; tb++) {
    int t0 = tb * 32;
    if (tb + 1 < nTB) {                 // prefetch next K/V block (keep near)
      __builtin_prefetch((const void*)(k + (long)(t0 + 32 + lane) * Dd), 0, 3);
      __builtin_prefetch((const void*)(v + (long)lane * Ss + t0 + 32), 0, 3);
      __builtin_prefetch((const void*)(v + (long)(lane + 32) * Ss + t0 + 32), 0, 3);
    }
    // S^T tiles: frag0 rows t0..t0+15, frag1 rows t0+16..t0+31 ; columns = queries
    f32x8 s0 = {}, s1 = {};
    {
      bf16x16 ka0 = load_frag_a(k, Dd, t0, 0, lane);
      bf16x16 ka1 = load_frag_a(k, Dd, t0, 32, lane);
      s0 = wmma_bf16(ka0, qb0, s0);
      s0 = wmma_bf16(ka1, qb1, s0);
      bf16x16 ka2 = load_frag_a(k, Dd, t0 + 16, 0, lane);
      bf16x16 ka3 = load_frag_a(k, Dd, t0 + 16, 32, lane);
      s1 = wmma_bf16(ka2, qb0, s1);
      s1 = wmma_bf16(ka3, qb1, s1);
    }
    // causal mask only on the diagonal block; 1/sqrt(S) already folded into Q
    float pm = mst;
    if (tb < nFull) {
#pragma unroll
      for (int r = 0; r < 8; r++) pm = fmaxf(pm, fmaxf(s0[r], s1[r]));
    } else {
      int tr0 = t0 + half * 8;          // frag0 row of slot 0
#pragma unroll
      for (int r = 0; r < 8; r++) {
        float va = (tr0 + r      <= m_col) ? s0[r] : -1e30f;
        float vb = (tr0 + r + 16 <= m_col) ? s1[r] : -1e30f;
        s0[r] = va; s1[r] = vb;
        pm = fmaxf(pm, fmaxf(va, vb));
      }
    }
    pm = fmaxf(pm, swap16(pm));         // combine the two half-wave key subsets
    float corr = __expf(mst - pm);
    mst = pm;
    float psum = 0.f;
#pragma unroll
    for (int r = 0; r < 8; r++) {
      float e0 = __expf(s0[r] - pm);
      float e1 = __expf(s1[r] - pm);
      s0[r] = e0; s1[r] = e1;
      psum += e0 + e1;
    }
    psum += swap16(psum);
    lst = lst * corr + psum;
    if (corr < 1.0f) {                  // skip rescale when max unchanged (execz fast path)
#pragma unroll
      for (int t = 0; t < 4; t++)
#pragma unroll
        for (int r = 0; r < 8; r++) o[t][r] *= corr;
    }

    // Build B-fragment of P^T (32x16: K=t, N=query) from the two score fragments.
    // lane<16 needs k=0..15 -> all of frag0 (own slots t0+r, partner t0+8+r)
    // lane>=16 needs k=16..31 -> all of frag1 (partner t0+16+r, own t0+24+r)
    bf16x16 pb;
#pragma unroll
    for (int r = 0; r < 8; r++) {
      float p0p = swap16(s0[r]);
      float p1p = swap16(s1[r]);
      float e_lo = half ? p1p   : s0[r];
      float e_hi = half ? s1[r] : p0p;
      pb[r]     = f2bfh(e_lo);
      pb[r + 8] = f2bfh(e_hi);
    }
    // O^T += V^T(16x32 per d-tile) @ P^T(32x16)
#pragma unroll
    for (int t = 0; t < 4; t++) {
      bf16x16 va = load_frag_a(v, Ss, t * 16, t0, lane);
      o[t] = wmma_bf16(va, pb, o[t]);
    }
  }

  // epilogue: per-lane normalize, packed 128-bit stores into hidden (B,S,E) bf16
  float inv = 1.f / lst;
  long rowBase = ((long)b * Ss + mw + col) * Ee + h * Dd;
#pragma unroll
  for (int t = 0; t < 4; t++) {
    u16x8 pk;
#pragma unroll
    for (int r = 0; r < 8; r++) pk[r] = f2bf(o[t][r] * inv);
    *reinterpret_cast<u16x8*>(Hid + rowBase + t * 16 + half * 8) = pk;
  }
}

// ---------- launcher ----------
extern "C" void kernel_launch(void* const* d_in, const int* in_sizes, int n_in,
                              void* d_out, int out_size, void* d_ws, size_t ws_size,
                              hipStream_t stream) {
  const float* x  = (const float*)d_in[0];
  const float* Wq = (const float*)d_in[1];
  const float* bq = (const float*)d_in[2];
  const float* Wk = (const float*)d_in[3];
  const float* bk = (const float*)d_in[4];
  const float* Wv = (const float*)d_in[5];
  const float* bv = (const float*)d_in[6];
  const float* Wo = (const float*)d_in[7];
  const float* bo = (const float*)d_in[8];
  float* out = (float*)d_out;

  unsigned short* ws = (unsigned short*)d_ws;
  const long nX = (long)Bb * Ss * Ee;
  const long nW = (long)Hh * Dd * Ee;
  const long nO = (long)Ee * Ee;
  const long nQ = (long)Bb * Hh * Ss * Dd;
  unsigned short* xb  = ws; ws += nX;
  unsigned short* wqT = ws; ws += nW;
  unsigned short* wkT = ws; ws += nW;
  unsigned short* wvT = ws; ws += nW;
  unsigned short* woT = ws; ws += nO;
  unsigned short* qb  = ws; ws += nQ;
  unsigned short* kb  = ws; ws += nQ;
  unsigned short* vT  = ws; ws += nQ;
  unsigned short* hid = ws; ws += nX;
  (void)ws_size; (void)in_sizes; (void)n_in; (void)out_size;

  // 1) precision staging: x -> bf16, weights -> transposed bf16 (N,K)
  k_f32_to_bf16<<<2048, 256, 0, stream>>>(x, xb, nX);
  k_transpose_bf16<<<dim3(256, 1, Hh), 256, 0, stream>>>(Wq, wqT, Ee, Dd);
  k_transpose_bf16<<<dim3(256, 1, Hh), 256, 0, stream>>>(Wk, wkT, Ee, Dd);
  k_transpose_bf16<<<dim3(256, 1, Hh), 256, 0, stream>>>(Wv, wvT, Ee, Dd);
  k_transpose_bf16<<<dim3(512, 1, 1), 256, 0, stream>>>(Wo, woT, Ee, Ee);

  // 2) QKV projections: per (b,h), (S x E) @ (E x D) + bias.
  //    Q gets the 1/sqrt(S)=1/32 softmax scale folded in (exact: power of two).
  dim3 gp(Ss / 256, 1, Bb * Hh);
  k_gemm<0><<<gp, 256, 0, stream>>>(xb, Ee, (long)Ss * Ee, Hh, wqT, Ee, (long)Dd * Ee,
                                    bq, Dd, qb, Dd, (long)Ss * Dd, Ee, 0.03125f);
  k_gemm<0><<<gp, 256, 0, stream>>>(xb, Ee, (long)Ss * Ee, Hh, wkT, Ee, (long)Dd * Ee,
                                    bk, Dd, kb, Dd, (long)Ss * Dd, Ee, 1.0f);
  k_gemm<1><<<gp, 256, 0, stream>>>(xb, Ee, (long)Ss * Ee, Hh, wvT, Ee, (long)Dd * Ee,
                                    bv, Dd, vT, Ss, (long)Dd * Ss, Ee, 1.0f);

  // 3) fused causal attention -> hidden (B,S,E) bf16
  k_flash<<<dim3(Ss / 128, Bb * Hh), 256, 0, stream>>>(qb, kb, vT, hid);

  // 4) output projection: (B*S x E) @ (E x E) + bo -> fp32 out
  dim3 go((Bb * Ss) / 256, Ee / 64, 1);
  k_gemm<2><<<go, 256, 0, stream>>>(hid, Ee, 0, 1, woT, Ee, 0,
                                    bo, 0, out, Ee, 0, Ee, 1.0f);
}
